// CrossregionRelationshipModeling_25864293056857
// MI455X (gfx1250) — matrix operand order repaired and checked by reading, not compiled, not run on hardware
//
#include <hip/hip_runtime.h>
#include <hip/hip_bf16.h>

typedef __attribute__((ext_vector_type(16))) __bf16 v16bf;
typedef __attribute__((ext_vector_type(8)))  float  v8f;

#define N_NODES   12
#define C_DIM     512
#define BT        4                       // batches per workgroup
#define ROWS      (BT * N_NODES)          // 48 rows of x per workgroup
#define KSTEPS    (C_DIM / 32)            // 16 k-steps of K=32 (bf16 WMMA)
#define AS_STRIDE 520                     // bf16 elems; 1040B rows: 16B aligned, banks spread
#define WT_STRIDE 40                      // bf16 elems; 80B rows: 16B aligned
#define H_STRIDE  516                     // fp32 elems; 2064B rows: 16B aligned
#define OFF_WT_BYTES    (ROWS * AS_STRIDE * 2)                 // 49920
#define WT_BUF_ELEMS    (C_DIM * WT_STRIDE)                    // 20480 bf16 / 40960B
#define OFF_SMALL_BYTES (OFF_WT_BYTES + 2 * WT_BUF_ELEMS * 2)  // 131840
#define SMEM_BYTES      (OFF_SMALL_BYTES + 2688)               // +el/er/alpha = 134528

// ---- CDNA5 async global->LDS copy (no VGPR round-trip, tracked by ASYNCcnt) ----
// INST_OFFSET applies to BOTH lds and global addresses (ISA 08_async_tensor 4.4),
// so one addr pair + offsets copies 64B contiguous on both sides.
// LDS operand: flat LDS aperture addr[31:0] == LDS byte address (ISA 10.2).
__device__ __forceinline__ void async_copy_64B(void* lds_dst, const void* gsrc) {
    unsigned ldsa = (unsigned)(unsigned long long)lds_dst;
    asm volatile(
        "global_load_async_to_lds_b128 %0, %1, off\n\t"
        "global_load_async_to_lds_b128 %0, %1, off offset:16\n\t"
        "global_load_async_to_lds_b128 %0, %1, off offset:32\n\t"
        "global_load_async_to_lds_b128 %0, %1, off offset:48"
        :: "v"(ldsa), "v"(gsrc)
        : "memory");
}
__device__ __forceinline__ void wait_async_all() {
    asm volatile("s_wait_asynccnt 0x0" ::: "memory");
}

// issue one K=32 chunk of Wt (512 cols x 64B) as async copies: 8 per thread
__device__ __forceinline__ void issue_wt_chunk(const __bf16* __restrict__ Wt,
                                               __bf16* dst, int chunk, int tid) {
    const __bf16* srcb = Wt + chunk * 32;
    const int c0 = tid;
    const int c1 = tid + 256;
    async_copy_64B(dst + c0 * WT_STRIDE, srcb + (size_t)c0 * C_DIM);
    async_copy_64B(dst + c1 * WT_STRIDE, srcb + (size_t)c1 * C_DIM);
}

// ---------- prep: W [K][N] fp32 -> Wt [N][K] bf16 (one-time, L2-resident) ----------
__global__ void __launch_bounds__(256) wt_prep_kernel(const float* __restrict__ W,
                                                      __bf16* __restrict__ Wt) {
    int idx = blockIdx.x * 256 + threadIdx.x;    // 512*512 total
    int k = idx >> 9;
    int c = idx & 511;
    Wt[c * C_DIM + k] = (__bf16)W[k * C_DIM + c];
}

// ---------- main GAT kernel ----------
__global__ void __launch_bounds__(256) gat_kernel(
    const float* __restrict__ x, const float* __restrict__ adj_mask,
    const __bf16* __restrict__ Wt, const float* __restrict__ a_l,
    const float* __restrict__ a_r, float* __restrict__ out)
{
    extern __shared__ __align__(16) unsigned char smem[];
    __bf16* As  = (__bf16*)smem;                          // [48][520] bf16 x tile
    __bf16* WtS = (__bf16*)(smem + OFF_WT_BYTES);         // 2 x [512][40] bf16 W chunks
    float*  elS = (float*)(smem + OFF_SMALL_BYTES);       // [48]
    float*  erS = elS + ROWS;                             // [48]
    float*  alS = erS + ROWS;                             // [4][12][12] alpha
    float*  hS  = (float*)smem;                           // [48][516] fp32 h (overlays As/WtS)

    const int tid  = threadIdx.x;
    const int lane = tid & 31;
    const int wave = tid >> 5;          // 8 waves
    const int half = lane >> 4;         // lane half (0/1) per WMMA layout
    const int ln16 = lane & 15;
    const size_t row_base = (size_t)blockIdx.x * ROWS;
    const float* xg = x + row_base * C_DIM;

    // ---- kick off async copy of Wt chunk 0 (overlaps with x-tile conversion) ----
    issue_wt_chunk(Wt, WtS, 0, tid);

    // ---- Phase A: load x tile (48x512 fp32), convert to bf16 in LDS ----
    for (int i = tid; i < ROWS * C_DIM / 4; i += 256) {
        int r = i >> 7;                   // (i*4) / 512
        int k = (i & 127) * 4;
        float4 v = *(const float4*)(xg + r * C_DIM + k);
        union { __bf16 h[4]; unsigned long long u; } pk;
        pk.h[0] = (__bf16)v.x; pk.h[1] = (__bf16)v.y;
        pk.h[2] = (__bf16)v.z; pk.h[3] = (__bf16)v.w;
        *(unsigned long long*)(As + r * AS_STRIDE + k) = pk.u;
    }
    wait_async_all();
    __syncthreads();

    // ---- Phase B: GEMM h = x @ W via bf16 WMMA, fp32 accumulate ----
    // wave owns columns [wave*64, wave*64+64): 3 M-tiles x 4 N-tiles accumulators
    v8f acc[3][4];
    #pragma unroll
    for (int mt = 0; mt < 3; ++mt)
        #pragma unroll
        for (int nt = 0; nt < 4; ++nt)
            acc[mt][nt] = (v8f){0.f, 0.f, 0.f, 0.f, 0.f, 0.f, 0.f, 0.f};

    const int col0 = wave * 64;
    for (int ks = 0; ks < KSTEPS; ++ks) {
        const __bf16* cur = WtS + (ks & 1) * WT_BUF_ELEMS;
        // async-prefetch next K-chunk of W into the other LDS buffer
        if (ks + 1 < KSTEPS)
            issue_wt_chunk(Wt, WtS + ((ks + 1) & 1) * WT_BUF_ELEMS, ks + 1, tid);

        // A fragments: 16x32 bf16, lane ln16 = row; K halves per ISA layout
        v16bf af[3];
        #pragma unroll
        for (int mt = 0; mt < 3; ++mt) {
            const __bf16* p = As + (mt * 16 + ln16) * AS_STRIDE + ks * 32 + half * 8;
            union { v16bf v; float4 f[2]; } u;
            u.f[0] = *(const float4*)p;           // K = half*8 + [0..7]
            u.f[1] = *(const float4*)(p + 16);    // K = 16 + half*8 + [0..7]
            af[mt] = u.v;
        }
        // B fragments: 32x16 bf16 from transposed Wt (contiguous K per column)
        v16bf bf[4];
        #pragma unroll
        for (int nt = 0; nt < 4; ++nt) {
            const __bf16* p = cur + (col0 + nt * 16 + ln16) * WT_STRIDE + half * 8;
            union { v16bf v; float4 f[2]; } u;
            u.f[0] = *(const float4*)p;           // K = half*8 + [0..7]
            u.f[1] = *(const float4*)(p + 16);    // K = 16 + half*8 + [0..7]
            bf[nt] = u.v;
        }
        #pragma unroll
        for (int mt = 0; mt < 3; ++mt)
            #pragma unroll
            for (int nt = 0; nt < 4; ++nt)
                acc[mt][nt] = __builtin_amdgcn_wmma_f32_16x16x32_bf16(
                    false, af[mt], false, bf[nt], (short)0, acc[mt][nt], false, false);

        wait_async_all();      // our chunk ks+1 copies are in LDS before the barrier
        __syncthreads();
    }

    // ---- Phase C: spill h (fp32) to LDS, overlaying As/WtS ----
    #pragma unroll
    for (int mt = 0; mt < 3; ++mt)
        #pragma unroll
        for (int nt = 0; nt < 4; ++nt) {
            int col = col0 + nt * 16 + ln16;
            int rb  = mt * 16 + half * 8;
            #pragma unroll
            for (int v = 0; v < 8; ++v)
                hS[(rb + v) * H_STRIDE + col] = acc[mt][nt][v];
        }
    __syncthreads();

    // ---- Phase D: el = h.a_l, er = h.a_r (wave-parallel row reductions) ----
    {
        float al[16], ar[16];
        #pragma unroll
        for (int i = 0; i < 16; ++i) {
            al[i] = a_l[i * 32 + lane];
            ar[i] = a_r[i * 32 + lane];
        }
        for (int rr = 0; rr < ROWS / 8; ++rr) {
            int r = wave * (ROWS / 8) + rr;
            float pl = 0.f, pr = 0.f;
            #pragma unroll
            for (int i = 0; i < 16; ++i) {
                float v = hS[r * H_STRIDE + i * 32 + lane];
                pl = fmaf(v, al[i], pl);
                pr = fmaf(v, ar[i], pr);
            }
            #pragma unroll
            for (int off = 16; off > 0; off >>= 1) {
                pl += __shfl_xor(pl, off, 32);
                pr += __shfl_xor(pr, off, 32);
            }
            if (lane == 0) { elS[r] = pl; erS[r] = pr; }
        }
    }
    __syncthreads();

    // ---- Phase E: masked softmax over neighbors (one thread per (batch,node)) ----
    if (tid < BT * N_NODES) {
        // ADJ_BASE rows as 12-bit masks; adj>0.1 <=> (m==n) || (base && mask>0.1)
        const unsigned short ADJROW[12] = {0xFE8, 0xFE8, 0xFE8, 0xFF7, 0xFE8, 0x1DF,
                                           0xE3F, 0xE3F, 0xE3F, 0x1DF, 0x1DF, 0x1DF};
        int bl = tid / N_NODES, n = tid - bl * N_NODES;
        const float* am = adj_mask + ((size_t)blockIdx.x * BT + bl) * (N_NODES * N_NODES)
                          + n * N_NODES;
        float eln = elS[bl * N_NODES + n];
        unsigned rowm = ADJROW[n];
        float ev[N_NODES];
        bool  vd[N_NODES];
        float mx = -3.402823e38f;
        #pragma unroll
        for (int m = 0; m < N_NODES; ++m) {
            bool valid = (m == n) || (((rowm >> m) & 1u) && (am[m] > 0.1f));
            float e = eln + erS[bl * N_NODES + m];
            e = (e > 0.f) ? e : 0.2f * e;          // leaky_relu
            vd[m] = valid; ev[m] = e;
            if (valid && e > mx) mx = e;
        }
        float s = 0.f;
        #pragma unroll
        for (int m = 0; m < N_NODES; ++m) {
            float w = vd[m] ? __expf(ev[m] - mx) : 0.f;
            ev[m] = w; s += w;
        }
        float inv = 1.f / s;
        #pragma unroll
        for (int m = 0; m < N_NODES; ++m)
            alS[bl * 144 + n * N_NODES + m] = ev[m] * inv;
    }
    __syncthreads();

    // ---- Phase F: out = alpha @ h + x (float4, LDS broadcast of alpha) ----
    float* og = out + row_base * C_DIM;
    for (int it = 0; it < ROWS * (C_DIM / 4) / 256; ++it) {
        int idx = tid + it * 256;
        int r   = idx >> 7;                 // row in [0,48)
        int c   = (idx & 127) * 4;          // column (float4)
        int bl  = r / N_NODES;
        int n   = r - bl * N_NODES;
        const float* av = alS + bl * 144 + n * N_NODES;
        float4 a4 = *(const float4*)(xg + r * C_DIM + c);   // residual (L2 hit)
        #pragma unroll
        for (int m = 0; m < N_NODES; ++m) {
            float w = av[m];
            float4 h4 = *(const float4*)(hS + (bl * N_NODES + m) * H_STRIDE + c);
            a4.x = fmaf(w, h4.x, a4.x);
            a4.y = fmaf(w, h4.y, a4.y);
            a4.z = fmaf(w, h4.z, a4.z);
            a4.w = fmaf(w, h4.w, a4.w);
        }
        *(float4*)(og + r * C_DIM + c) = a4;
    }
}

extern "C" void kernel_launch(void* const* d_in, const int* in_sizes, int n_in,
                              void* d_out, int out_size, void* d_ws, size_t ws_size,
                              hipStream_t stream) {
    const float* x  = (const float*)d_in[0];
    const float* am = (const float*)d_in[1];
    const float* W  = (const float*)d_in[2];
    const float* al = (const float*)d_in[3];
    const float* ar = (const float*)d_in[4];
    __bf16* Wt = (__bf16*)d_ws;                 // 512KB bf16 transposed W

    // allow >64KB dynamic LDS (idempotent, host-side, capture-safe)
    (void)hipFuncSetAttribute((const void*)gat_kernel,
                              hipFuncAttributeMaxDynamicSharedMemorySize, SMEM_BYTES);

    wt_prep_kernel<<<(C_DIM * C_DIM) / 256, 256, 0, stream>>>(W, Wt);

    const int B = in_sizes[0] / (N_NODES * C_DIM);   // 16384
    gat_kernel<<<B / BT, 256, SMEM_BYTES, stream>>>(x, am, Wt, al, ar, (float*)d_out);
}
